// DeconvBlock_88261577933061
// MI455X (gfx1250) — compile-verified, matrix operand order checked by reading, not run on hardware
//
#include <hip/hip_runtime.h>
#include <cstdint>

// ---------------------------------------------------------------------------
// Problem constants (from reference)
// ---------------------------------------------------------------------------
#define BB    16
#define CIN   256
#define COUT  128
#define HH    64
#define WW    64
#define KK    4
#define OH    128
#define OW    128

// weights (bf16 hi/lo planes) element count per plane
#define TOTW  (BB * 16 * CIN * COUT)   // 8,388,608

typedef __attribute__((ext_vector_type(16))) __bf16        v16bf;
typedef __attribute__((ext_vector_type(8)))  float         v8f;
typedef __attribute__((ext_vector_type(4)))  unsigned int  v4u;

union AFrag { uint32_t d[8];  v16bf v; };
union BFrag { v4u      q[2];  v16bf v; };

__device__ __forceinline__ unsigned short f32_to_bf16_rn(float f) {
    uint32_t u = __float_as_uint(f);
    u += 0x7FFFu + ((u >> 16) & 1u);          // round-to-nearest-even
    return (unsigned short)(u >> 16);
}
__device__ __forceinline__ float bf16_to_f32(unsigned short h) {
    return __uint_as_float(((uint32_t)h) << 16);
}

// ---------------------------------------------------------------------------
// Kernel 1: synthesize per-sample weights, split fp32 -> bf16 hi/lo, and store
// pre-swizzled in WMMA B-fragment order:
//   frag = ((b*16 + tap)*8 + ci_chunk)*8 + co_tile      (512 elems per frag)
//   within frag: lane*16 + h   with lane = ncol + 16*khalf,
//   kloc = ci&31:  khalf=(kloc>>3)&1,  h = (kloc&7) | (((kloc>>4)&1)<<3)
// ---------------------------------------------------------------------------
__global__ __launch_bounds__(256) void synth_weights_kernel(
    const float* __restrict__ weight,
    const float* __restrict__ feature,
    const float* __restrict__ tb, const float* __restrict__ tq,
    const float* __restrict__ tn, const float* __restrict__ tx,
    const float* __restrict__ mb, const float* __restrict__ mq,
    const float* __restrict__ mn, const float* __restrict__ mx,
    unsigned short* __restrict__ wsb)
{
    int tid = blockIdx.x * 256 + threadIdx.x;
    if (tid >= TOTW) return;

    int co  = tid & (COUT - 1);
    int ci  = (tid >> 7) & (CIN - 1);
    int tap = (tid >> 15) & 15;
    int b   = tid >> 19;
    int ky  = tap >> 2;
    int kx  = tap & 3;

    int widx = ((ci * COUT + co) * KK + ky) * KK + kx;  // (CIN,COUT,K,K)
    int midx = ci * COUT + co;                          // (CIN,COUT,1,1)
    const float* f = feature + b * 4;

    float w = weight[widx]
            + tb[widx] * mb[midx] * f[0]
            + tq[widx] * mq[midx] * f[1]
            + tn[widx] * mn[midx] * f[2]
            + tx[widx] * mx[midx] * f[3];

    unsigned short hi  = f32_to_bf16_rn(w);
    float          rem = w - bf16_to_f32(hi);
    unsigned short lo  = f32_to_bf16_rn(rem);

    int kloc  = ci & 31;
    int ncol  = co & 15;
    int khalf = (kloc >> 3) & 1;
    int h     = (kloc & 7) | (((kloc >> 4) & 1) << 3);
    int lane  = ncol + 16 * khalf;
    size_t frag = (size_t)((b * 16 + tap) * 8 + (ci >> 5)) * 8 + (co >> 4);
    size_t off  = frag * 512 + lane * 16 + h;

    wsb[off]        = hi;
    wsb[TOTW + off] = lo;
}

// ---------------------------------------------------------------------------
// Kernel 2: transposed conv as parity-split GEMM with bf16x3 WMMA.
// Block = 256 threads = 8 waves; each wave owns one 16-cout tile.
// One block covers: sample b, output rows (oy0, oy0+1), 32 output columns
// (both ox parities). All four (py,px) parity jobs share one LDS-staged
// input patch: 256 ci x 3 rows x 18 cols, as bf16 hi/lo.
// ---------------------------------------------------------------------------
#define LPITCH 258              // ci pitch (padded, even) to avoid bank camps
#define LCOLS  18               // column halo: ix in [16*bx-1, 16*bx+16]
#define LROWS  3                // iy in [by-1, by+1]
#define LRC    (LROWS * LCOLS)  // 54

__global__ __launch_bounds__(256) void deconv_wmma_kernel(
    const float* __restrict__ x,
    const unsigned short* __restrict__ wsb,
    const float* __restrict__ bias,
    const float* __restrict__ prelu,
    float* __restrict__ out)
{
    __shared__ unsigned short lhsH[LRC * LPITCH];
    __shared__ unsigned short lhsL[LRC * LPITCH];

    const int bx = blockIdx.x;            // 32-col output block: ox in [32bx, 32bx+31]
    const int by = blockIdx.y;            // output row pair: oy in {2by, 2by+1}
    const int b  = blockIdx.z;

    const int xq0    = bx * 16;           // per-parity column base
    const int iybase = by - 1;
    const int ixbase = xq0 - 1;

    // ---- stage input patch into LDS as bf16 hi/lo -------------------------
    const float* xb = x + (size_t)b * CIN * HH * WW;
    for (int idx = threadIdx.x; idx < CIN * LRC; idx += 256) {
        int ci = idx / LRC;
        int rc = idx - ci * LRC;          // rc = r*18 + cc
        int r  = rc / LCOLS;
        int cc = rc - r * LCOLS;
        int iy = iybase + r;
        int ix = ixbase + cc;
        float v = 0.0f;
        if (iy >= 0 && iy < HH && ix >= 0 && ix < WW)
            v = xb[(ci * HH + iy) * WW + ix];
        unsigned short hi  = f32_to_bf16_rn(v);
        float          rem = v - bf16_to_f32(hi);
        unsigned short lo  = f32_to_bf16_rn(rem);
        lhsH[rc * LPITCH + ci] = hi;
        lhsL[rc * LPITCH + ci] = lo;
    }
    __syncthreads();

    // ---- per-wave 16(M) x 16(N) tiles, K = 256 over 4 taps, 4 parity jobs -
    const int lane  = threadIdx.x & 31;
    const int wave  = threadIdx.x >> 5;
    const int n0    = wave * 16;          // 8 waves cover all 128 couts
    const int mrow  = lane & 15;
    const int khalf = lane >> 4;

    const int   ncol = lane & 15;
    const int   co   = n0 + ncol;
    const float bsv  = bias[co];
    const float aco  = prelu[0];

    #pragma unroll 1
    for (int job = 0; job < 4; ++job) {
        const int pj = job >> 1;          // oy parity (oy = 2by + pj)
        const int px = job & 1;           // ox parity
        const int kyB = 1 - pj;
        const int kxB = 1 - px;

        v8f acc;
        #pragma unroll
        for (int i = 0; i < 8; ++i) acc[i] = 0.0f;

        #pragma unroll
        for (int dy = 0; dy < 2; ++dy) {
            #pragma unroll
            for (int dx = 0; dx < 2; ++dx) {
                const int ky  = kyB + 2 * dy;
                const int kx  = kxB + 2 * dx;
                const int tap = ky * 4 + kx;
                const int r   = 1 + pj - dy;              // LDS row
                const int cc  = mrow + 1 - dx + px;       // LDS column for this M
                const int rcb = (r * LCOLS + cc) * LPITCH;

                #pragma unroll
                for (int chunk = 0; chunk < 8; ++chunk) {
                    const int ci0 = chunk * 32;

                    // A fragments from LDS (documented 16-bit A layout)
                    AFrag ah, al;
                    #pragma unroll
                    for (int vg = 0; vg < 8; ++vg) {
                        int K0 = ((2 * vg) & 7) + 8 * khalf + ((vg >= 4) ? 16 : 0);
                        int li = rcb + ci0 + K0;          // even -> dword aligned
                        ah.d[vg] = *(const uint32_t*)(&lhsH[li]);
                        al.d[vg] = *(const uint32_t*)(&lhsL[li]);
                    }

                    // B fragments: pre-swizzled, two 16B vector loads per lane
                    size_t fb = ((size_t)((b * 16 + tap) * 8 + chunk) * 8
                                 + (n0 >> 4)) * 512 + (size_t)lane * 16;
                    const v4u* ph = (const v4u*)(wsb + fb);
                    const v4u* pl = (const v4u*)(wsb + TOTW + fb);
                    BFrag bh, bl;
                    bh.q[0] = ph[0]; bh.q[1] = ph[1];
                    bl.q[0] = pl[0]; bl.q[1] = pl[1];

                    // prefetch next K-chunk of B (chunk stride = 8 frags)
                    __builtin_prefetch((const void*)(wsb + fb + 4096), 0, 1);

                    // bf16x3: ah*bh + ah*bl + al*bh  (~fp32 accuracy)
                    acc = __builtin_amdgcn_wmma_f32_16x16x32_bf16(
                              false, ah.v, false, bh.v, (short)0, acc, false, false);
                    acc = __builtin_amdgcn_wmma_f32_16x16x32_bf16(
                              false, ah.v, false, bl.v, (short)0, acc, false, false);
                    acc = __builtin_amdgcn_wmma_f32_16x16x32_bf16(
                              false, al.v, false, bh.v, (short)0, acc, false, false);
                }
            }
        }

        // ---- epilogue: bias + PReLU, scatter to NCHW output ---------------
        const int oy = 2 * by + pj;
        float* ob = out + ((size_t)(b * COUT + co) * OH + oy) * OW;

        #pragma unroll
        for (int rr = 0; rr < 8; ++rr) {
            int M  = rr + 8 * khalf;      // C layout: M = vgpr + 8*(lane>>4)
            int ox = 2 * (xq0 + M) + px;
            float v = acc[rr] + bsv;
            ob[ox] = (v >= 0.0f) ? v : aco * v;
        }
    }
}

// ---------------------------------------------------------------------------
// Host launcher
// ---------------------------------------------------------------------------
extern "C" void kernel_launch(void* const* d_in, const int* in_sizes, int n_in,
                              void* d_out, int out_size, void* d_ws, size_t ws_size,
                              hipStream_t stream)
{
    const float* x       = (const float*)d_in[0];
    const float* feature = (const float*)d_in[1];
    const float* weight  = (const float*)d_in[2];
    const float* t_bayer = (const float*)d_in[3];
    const float* t_quad  = (const float*)d_in[4];
    const float* t_nano  = (const float*)d_in[5];
    const float* t_qxq   = (const float*)d_in[6];
    const float* m_bayer = (const float*)d_in[7];
    const float* m_quad  = (const float*)d_in[8];
    const float* m_nano  = (const float*)d_in[9];
    const float* m_qxq   = (const float*)d_in[10];
    const float* bias    = (const float*)d_in[11];
    const float* prelu_a = (const float*)d_in[12];
    float* out = (float*)d_out;
    unsigned short* wsb = (unsigned short*)d_ws;

    // 1) synthesize + swizzle weights (bf16 hi/lo planes in workspace)
    {
        int total  = TOTW;
        int blocks = (total + 255) / 256;
        synth_weights_kernel<<<blocks, 256, 0, stream>>>(
            weight, feature, t_bayer, t_quad, t_nano, t_qxq,
            m_bayer, m_quad, m_nano, m_qxq, wsb);
    }

    // 2) parity-merged transposed conv via WMMA
    {
        dim3 grid(4, OH / 2, BB);   // 32-col blocks x row pairs x batch
        deconv_wmma_kernel<<<grid, 256, 0, stream>>>(x, wsb, bias, prelu_a, out);
    }
}
